// CrossAttention_47519518163387
// MI455X (gfx1250) — compile-verified, matrix operand order checked by reading, not run on hardware
//
#include <hip/hip_runtime.h>
#include <hip/hip_bf16.h>

// CDNA5 / gfx1250: wave32, WMMA 16x16x32 f16 -> f32 accumulate.
typedef __attribute__((ext_vector_type(16))) _Float16 v16h;
typedef __attribute__((ext_vector_type(8)))  _Float16 v8h;
typedef __attribute__((ext_vector_type(8)))  float    v8f;
typedef __attribute__((ext_vector_type(4)))  float    v4f;

#define B_      16
#define T_      4096
#define S_      77
#define DC_     768
#define H_      8
#define C_      40
#define INNER_  320
#define SPAD_N  80   // padded S for score N-dim (5 tiles of 16)
#define SPAD_K  96   // padded S for attn@V K-dim (3 k-steps of 32)
#define CPAD_K  64   // padded C for q.k K-dim   (2 k-steps of 32)
#define CPAD_N  48   // padded C for attn@V N-dim (3 tiles of 16)
#define SCALE_F 0.3976353643835253f   // 40^(-1/4), applied to q and to k

// 16-bit A/B fragment layout (CDNA5 ISA 7.12.2, 16x32):
// lane 0-15 : elements 0..7 = K 0..7,  elements 8..15 = K 16..23
// lane 16-31: elements 0..7 = K 8..15, elements 8..15 = K 24..31
// => per lane: two contiguous 8-half (16 B) chunks at K = base and base+16.
__device__ __forceinline__ v16h load_frag(const _Float16* p, int lane) {
  const int base = (lane & 16) ? 8 : 0;
  v8h lo = *(const v8h*)(p + base);        // K base .. base+7
  v8h hi = *(const v8h*)(p + base + 16);   // K base+16 .. base+23
  v16h r;
#pragma unroll
  for (int i = 0; i < 8; ++i) { r[i] = lo[i]; r[i + 8] = hi[i]; }
  return r;
}

// Same, but source is f32 (4x b128 loads + pack-convert).
__device__ __forceinline__ v16h load_frag_f32(const float* p, int lane) {
  const int base = (lane & 16) ? 8 : 0;
  v4f a0 = *(const v4f*)(p + base);
  v4f a1 = *(const v4f*)(p + base + 4);
  v4f b0 = *(const v4f*)(p + base + 16);
  v4f b1 = *(const v4f*)(p + base + 20);
  v16h r;
#pragma unroll
  for (int i = 0; i < 4; ++i) {
    r[i]      = (_Float16)a0[i];
    r[i + 4]  = (_Float16)a1[i];
    r[i + 8]  = (_Float16)b0[i];
    r[i + 12] = (_Float16)b1[i];
  }
  return r;
}

__device__ __forceinline__ v8f wmma16(v16h a, v16h b, v8f c) {
  return __builtin_amdgcn_wmma_f32_16x16x32_f16(
      /*neg_a=*/false, a, /*neg_b=*/false, b,
      /*c_mod=*/(short)0, c, /*reuse_a=*/false, /*reuse_b=*/false);
}

// ---------------------------------------------------------------------------
// Kernel 0a: WqT[h][c<48][d<320] = f16(Wq[d][h*40+c]), zero rows for c>=40.
// ---------------------------------------------------------------------------
__global__ void k_prep_wq(const float* __restrict__ Wq, _Float16* __restrict__ WqT) {
  int i = blockIdx.x * blockDim.x + threadIdx.x;           // h*48*320 + c*320 + d
  if (i >= H_ * CPAD_N * INNER_) return;
  int d = i % INNER_;
  int c = (i / INNER_) % CPAD_N;
  int h = i / (INNER_ * CPAD_N);
  WqT[i] = (c < C_) ? (_Float16)Wq[(size_t)d * INNER_ + h * C_ + c] : (_Float16)0.f;
}

// ---------------------------------------------------------------------------
// Kernel 0b: WoT[n][d] = f16(Wo[d][n])
// ---------------------------------------------------------------------------
__global__ void k_prep_wo(const float* __restrict__ Wo, _Float16* __restrict__ WoT) {
  int i = blockIdx.x * blockDim.x + threadIdx.x;           // n*320 + d
  if (i >= INNER_ * INNER_) return;
  int d = i % INNER_, n = i / INNER_;
  WoT[i] = (_Float16)Wo[(size_t)d * INNER_ + n];
}

// ---------------------------------------------------------------------------
// Kernel 1: K/V projection (context @ Wk|Wv), zero-padded f16 tiles.
// Kbuf [b][h][s<80][c<64] = SCALE * ctx.Wk   (K-contig along c)
// VbufT[b][h][c<48][s<96] =         ctx.Wv   (K-contig along s)
// One block per (s, b); context row staged in LDS.
// ---------------------------------------------------------------------------
__global__ void __launch_bounds__(256)
k_proj_kv(const float* __restrict__ ctx, const float* __restrict__ Wk,
          const float* __restrict__ Wv, _Float16* __restrict__ Kbuf,
          _Float16* __restrict__ VbufT) {
  __shared__ float row[DC_];
  const int s = blockIdx.x;          // 0..95
  const int b = blockIdx.y;          // 0..15
  const int tid = threadIdx.x;
  const bool valid = (s < S_);
  if (valid) {
    for (int i = tid; i < DC_; i += 256)
      row[i] = ctx[((size_t)b * S_ + s) * DC_ + i];
  }
  __syncthreads();
  const int NOUT = H_ * (CPAD_K + CPAD_N);  // 896 outputs per (b,s)
  for (int idx = tid; idx < NOUT; idx += 256) {
    if (idx < H_ * CPAD_K) {                // K element
      int h = idx >> 6, c = idx & 63;
      float v = 0.f;
      if (valid && c < C_) {
        int col = h * C_ + c;
        for (int d = 0; d < DC_; ++d) v += row[d] * Wk[d * INNER_ + col];
        v *= SCALE_F;
      }
      if (s < SPAD_N)
        Kbuf[(((size_t)b * H_ + h) * SPAD_N + s) * CPAD_K + c] = (_Float16)v;
    } else {                                // V element (transposed store)
      int j = idx - H_ * CPAD_K;
      int h = j / CPAD_N, c = j % CPAD_N;
      float v = 0.f;
      if (valid && c < C_) {
        int col = h * C_ + c;
        for (int d = 0; d < DC_; ++d) v += row[d] * Wv[d * INNER_ + col];
      }
      VbufT[(((size_t)b * H_ + h) * CPAD_N + c) * SPAD_K + s] = (_Float16)v;
    }
  }
}

// ---------------------------------------------------------------------------
// Kernel 2: fused Q-projection + scores + softmax + attn@V. One wave per
// (b, h, 16-row tile of T). Scores never leave LDS.
// ---------------------------------------------------------------------------
__global__ void __launch_bounds__(32)
k_attn(const float* __restrict__ x, const _Float16* __restrict__ WqT,
       const _Float16* __restrict__ Kbuf, const _Float16* __restrict__ VbufT,
       _Float16* __restrict__ AO) {
  const int lane = threadIdx.x;
  const int m    = lane & 15;
  const int rb   = (lane & 16) ? 8 : 0;     // D-layout row base
  const int t0   = blockIdx.x * 16;
  const int h    = blockIdx.y;
  const int b    = blockIdx.z;

  __shared__ __align__(16) _Float16 qs[16 * CPAD_K];  // q tile, f16, zero-padded
  __shared__ __align__(16) float    sc[16 * SPAD_N];  // score tile
  __shared__ __align__(16) _Float16 at[16 * SPAD_K];  // softmaxed attn, padded

  const v8f vzero = {0.f, 0.f, 0.f, 0.f, 0.f, 0.f, 0.f, 0.f};

  // ---- Q = x[b, t0:t0+16, :] @ Wq[:, h-head]  (48 padded cols) ----
  v8f qacc[3];
  qacc[0] = vzero; qacc[1] = vzero; qacc[2] = vzero;
  const float*    xrow = x + ((size_t)b * T_ + t0 + m) * INNER_;
  const _Float16* wq   = WqT + (size_t)h * CPAD_N * INNER_;
  for (int kk = 0; kk < INNER_ / 32; ++kk) {   // 10 k-steps
    const int d0 = kk * 32;
    v16h a = load_frag_f32(xrow + d0, lane);
#pragma unroll
    for (int jn = 0; jn < 3; ++jn) {
      v16h bf = load_frag(wq + (jn * 16 + m) * INNER_ + d0, lane);
      qacc[jn] = wmma16(a, bf, qacc[jn]);
    }
  }
  // scatter q (scaled) to LDS in f16, zero padding cols 40..63
  for (int i = lane; i < 16 * CPAD_K; i += 32) qs[i] = (_Float16)0.f;
  __syncthreads();
#pragma unroll
  for (int jn = 0; jn < 3; ++jn)
#pragma unroll
    for (int r = 0; r < 8; ++r) {
      const int col = jn * 16 + m;
      if (col < C_) qs[(r + rb) * CPAD_K + col] = (_Float16)(qacc[jn][r] * SCALE_F);
    }
  __syncthreads();

  // ---- scores[16 x 80] = q[16 x 64] @ K^T ----
  v8f sacc[5];
#pragma unroll
  for (int j = 0; j < 5; ++j) sacc[j] = vzero;
  const _Float16* kb = Kbuf + ((size_t)b * H_ + h) * SPAD_N * CPAD_K;
  for (int kk = 0; kk < 2; ++kk) {
    v16h a = load_frag(qs + m * CPAD_K + kk * 32, lane);
#pragma unroll
    for (int js = 0; js < 5; ++js) {
      v16h bf = load_frag(kb + (js * 16 + m) * CPAD_K + kk * 32, lane);
      sacc[js] = wmma16(a, bf, sacc[js]);
    }
  }
#pragma unroll
  for (int js = 0; js < 5; ++js)
#pragma unroll
    for (int r = 0; r < 8; ++r)
      sc[(r + rb) * SPAD_N + js * 16 + m] = sacc[js][r];
  __syncthreads();

  // ---- softmax over s<77, one row per lane 0..15; write f16 attn (pad 96) ----
  if (lane < 16) {
    float mx = -3.0e38f;
    for (int s = 0; s < S_; ++s) mx = fmaxf(mx, sc[lane * SPAD_N + s]);
    float sum = 0.f;
    for (int s = 0; s < S_; ++s) {
      float e = __expf(sc[lane * SPAD_N + s] - mx);
      sc[lane * SPAD_N + s] = e;
      sum += e;
    }
    const float inv = 1.f / sum;
    for (int s = 0; s < SPAD_K; ++s)
      at[lane * SPAD_K + s] =
          (s < S_) ? (_Float16)(sc[lane * SPAD_N + s] * inv) : (_Float16)0.f;
  }
  __syncthreads();

  // ---- out[16 x 48] = attn[16 x 96] @ V[96 x 48] (V stored transposed) ----
  v8f oacc[3];
  oacc[0] = vzero; oacc[1] = vzero; oacc[2] = vzero;
  const _Float16* vbt = VbufT + ((size_t)b * H_ + h) * CPAD_N * SPAD_K;
  for (int kk = 0; kk < 3; ++kk) {
    v16h a = load_frag(at + m * SPAD_K + kk * 32, lane);
#pragma unroll
    for (int jc = 0; jc < 3; ++jc) {
      v16h bf = load_frag(vbt + (jc * 16 + m) * SPAD_K + kk * 32, lane);
      oacc[jc] = wmma16(a, bf, oacc[jc]);
    }
  }
#pragma unroll
  for (int jc = 0; jc < 3; ++jc)
#pragma unroll
    for (int r = 0; r < 8; ++r) {
      const int col = jc * 16 + m;
      if (col < C_)
        AO[((size_t)b * T_ + t0 + r + rb) * INNER_ + h * C_ + col] =
            (_Float16)oacc[jc][r];
    }
}

// ---------------------------------------------------------------------------
// Kernel 3: out = AO(f16)[65536 x 320] @ Wo + bo, f32 out. Wo stored
// transposed f16 [n][d] so B fragments are K-contiguous. One wave per 16x64.
// ---------------------------------------------------------------------------
__global__ void __launch_bounds__(32)
k_outproj(const _Float16* __restrict__ AO, const _Float16* __restrict__ WoT,
          const float* __restrict__ bo, float* __restrict__ out) {
  const int lane = threadIdx.x;
  const int m    = lane & 15;
  const int rb   = (lane & 16) ? 8 : 0;
  const int m0   = blockIdx.x * 16;
  const int n0   = blockIdx.y * 64;

  const v8f vzero = {0.f, 0.f, 0.f, 0.f, 0.f, 0.f, 0.f, 0.f};
  v8f acc[4];
#pragma unroll
  for (int j = 0; j < 4; ++j) acc[j] = vzero;

  const _Float16* arow = AO + (size_t)(m0 + m) * INNER_;
  for (int kk = 0; kk < INNER_ / 32; ++kk) {   // 10 k-steps
    const int d0 = kk * 32;
    v16h a = load_frag(arow + d0, lane);
#pragma unroll
    for (int jn = 0; jn < 4; ++jn) {
      v16h bf = load_frag(WoT + (size_t)(n0 + jn * 16 + m) * INNER_ + d0, lane);
      acc[jn] = wmma16(a, bf, acc[jn]);
    }
  }
#pragma unroll
  for (int jn = 0; jn < 4; ++jn)
#pragma unroll
    for (int r = 0; r < 8; ++r) {
      const int col = n0 + jn * 16 + m;
      out[(size_t)(m0 + r + rb) * INNER_ + col] = acc[jn][r] + bo[col];
    }
}

// ---------------------------------------------------------------------------
extern "C" void kernel_launch(void* const* d_in, const int* in_sizes, int n_in,
                              void* d_out, int out_size, void* d_ws, size_t ws_size,
                              hipStream_t stream) {
  const float* x   = (const float*)d_in[0];
  const float* ctx = (const float*)d_in[1];
  const float* Wq  = (const float*)d_in[2];
  const float* Wk  = (const float*)d_in[3];
  const float* Wv  = (const float*)d_in[4];
  const float* Wo  = (const float*)d_in[5];
  const float* bo  = (const float*)d_in[6];
  float* out = (float*)d_out;

  // workspace carving (all offsets 256B aligned):
  //   WqT  : 8*48*320 f16          =    245,760 B
  //   WoT  : 320*320 f16           =    204,800 B
  //   Kbuf : 16*8*80*64 f16        =  1,310,720 B
  //   VbufT: 16*8*48*96 f16        =  1,179,648 B
  //   AO   : 65536*320 f16         = 41,943,040 B   (total ~44.9 MB)
  char* ws = (char*)d_ws;
  _Float16* WqT   = (_Float16*)(ws);
  _Float16* WoT   = (_Float16*)(ws + 245760);
  _Float16* Kbuf  = (_Float16*)(ws + 245760 + 204800);
  _Float16* VbufT = (_Float16*)(ws + 245760 + 204800 + 1310720);
  _Float16* AO    = (_Float16*)(ws + 245760 + 204800 + 1310720 + 1179648);

  k_prep_wq<<<(H_ * CPAD_N * INNER_ + 255) / 256, 256, 0, stream>>>(Wq, WqT);
  k_prep_wo<<<(INNER_ * INNER_ + 255) / 256, 256, 0, stream>>>(Wo, WoT);
  k_proj_kv<<<dim3(SPAD_K, B_), 256, 0, stream>>>(ctx, Wk, Wv, Kbuf, VbufT);
  k_attn<<<dim3(T_ / 16, H_, B_), 32, 0, stream>>>(x, WqT, Kbuf, VbufT, AO);
  k_outproj<<<dim3((B_ * T_) / 16, INNER_ / 64), 32, 0, stream>>>(AO, WoT, bo, out);
}
